// VQuantLinear_24103356465280
// MI455X (gfx1250) — compile-verified, matrix operand order checked by reading, not run on hardware
//
#include <hip/hip_runtime.h>

// ---------------------------------------------------------------------------
// VPTQ VQuantLinear for MI455X (gfx1250, wave32, WMMA)
//   Phase 1a: dequantize W (4096x4096) -> bf16 hi/lo planes in d_ws
//   Phase 1b: split x (16384x4096 f32)  -> bf16 hi/lo planes in d_ws
//   Phase 2 : y = x @ W^T via v_wmma_f32_16x16x32_bf16 with split-bf16
//             (x_hi*w_hi + x_lo*w_hi + x_hi*w_lo) ~ fp32 accuracy.
//             All tiles staged with global_load_async_to_lds_b128
//             (ASYNCcnt) into a double-buffered LDS pipeline.
// ---------------------------------------------------------------------------

typedef __attribute__((ext_vector_type(16))) __bf16 v16bf;
typedef __attribute__((ext_vector_type(8)))  __bf16 v8bf;
typedef __attribute__((ext_vector_type(4)))  __bf16 v4bf;
typedef __attribute__((ext_vector_type(8)))  float  v8f;
typedef __attribute__((ext_vector_type(4)))  float  v4f;
typedef __attribute__((ext_vector_type(4)))  int    v4i;
typedef __attribute__((ext_vector_type(2)))  unsigned v2u;

#define IN_F   4096   // in_features  (K)
#define OUT_F  4096   // out_features (N)
#define MTOT   16384  // batch*seq    (M) = 4*4096
#define VLEN   8

// round-to-nearest-even f32 -> bf16 (used in dequant only)
static __device__ __forceinline__ __bf16 f2bf(float f) {
  unsigned u = __builtin_bit_cast(unsigned, f);
  u += 0x7FFFu + ((u >> 16) & 1u);
  unsigned short s = (unsigned short)(u >> 16);
  return __builtin_bit_cast(__bf16, s);
}
static __device__ __forceinline__ float bf2f(__bf16 b) {
  unsigned short s = __builtin_bit_cast(unsigned short, b);
  unsigned u = ((unsigned)s) << 16;
  return __builtin_bit_cast(float, u);
}

// gfx1250 async global -> LDS copy, 16 bytes per lane, tracked by ASYNCcnt.
// lds_off = wave-relative LDS byte address (flat shared addr low 32 bits).
static __device__ __forceinline__ void async_load_b128(unsigned lds_off,
                                                       const void* gaddr) {
  asm volatile("global_load_async_to_lds_b128 %0, %1, off"
               :: "v"(lds_off), "v"(gaddr) : "memory");
}
// Async loads complete in order: waiting to <=6 outstanding guarantees the
// previous tile's 6 per-thread transfers have landed in LDS.
static __device__ __forceinline__ void wait_asynccnt6() {
  asm volatile("s_wait_asynccnt 0x6" ::: "memory");
}
static __device__ __forceinline__ void wait_asynccnt0() {
  asm volatile("s_wait_asynccnt 0x0" ::: "memory");
}
static __device__ __forceinline__ unsigned lds_off32(const void* p) {
  return (unsigned)(unsigned long long)p;   // LDS address = low 32 bits of flat addr
}

// ---------------------------------------------------------------------------
// Phase 1a: W[o,k] = (centroids[idx[o/8,k]*8 + o%8]
//                    + res_centroids[ridx[o/8,k]*8 + o%8]) * scale[k] + bias[k]
// split into bf16 hi/lo planes.  4 k's per thread (vector idx loads/stores).
// ---------------------------------------------------------------------------
__global__ void __launch_bounds__(256)
vptq_dequant_kernel(const float* __restrict__ centroids,
                    const int*   __restrict__ indices,
                    const float* __restrict__ res_centroids,
                    const int*   __restrict__ res_indices,
                    const float* __restrict__ wscale,
                    const float* __restrict__ wbias,
                    __bf16* __restrict__ w_hi,
                    __bf16* __restrict__ w_lo)
{
  unsigned t    = blockIdx.x * 256u + threadIdx.x;
  unsigned base = t * 4u;                 // flat element index into 4096x4096
  unsigned o    = base >> 12;             // output row (0..4095)
  unsigned k    = base & 4095u;           // input col, multiple of 4
  unsigned ni   = o >> 3;                 // vector-quant group
  unsigned v    = o & 7u;                 // position within VLEN

  const unsigned ib = ni * (unsigned)IN_F + k;
  v4i idx4  = *(const v4i*)(indices     + ib);
  v4i ridx4 = *(const v4i*)(res_indices + ib);
  v4f sc    = *(const v4f*)(wscale + k);
  v4f bi    = *(const v4f*)(wbias  + k);

  v4bf hi4, lo4;
#pragma unroll
  for (int j = 0; j < 4; ++j) {
    float w = centroids[(unsigned)idx4[j] * VLEN + v]
            + res_centroids[(unsigned)ridx4[j] * VLEN + v];
    w = w * sc[j] + bi[j];
    __bf16 h = f2bf(w);
    hi4[j] = h;
    lo4[j] = f2bf(w - bf2f(h));
  }
  size_t go = (size_t)o * IN_F + k;
  *(v4bf*)(w_hi + go) = hi4;
  *(v4bf*)(w_lo + go) = lo4;
}

// ---------------------------------------------------------------------------
// Phase 1b: split x into bf16 hi (truncated) / lo (exact residual, truncated)
// planes.  One pass, 4 elements per thread, b64 stores.
// ---------------------------------------------------------------------------
__global__ void __launch_bounds__(256)
vptq_xsplit_kernel(const float* __restrict__ x,
                   __bf16* __restrict__ x_hi,
                   __bf16* __restrict__ x_lo)
{
  size_t base = ((size_t)blockIdx.x * 256u + threadIdx.x) * 4u;
  v4f xv = *(const v4f*)(x + base);

  unsigned u0 = __builtin_bit_cast(unsigned, xv[0]);
  unsigned u1 = __builtin_bit_cast(unsigned, xv[1]);
  unsigned u2 = __builtin_bit_cast(unsigned, xv[2]);
  unsigned u3 = __builtin_bit_cast(unsigned, xv[3]);
  v2u hp;
  hp[0] = (u1 & 0xffff0000u) | (u0 >> 16);
  hp[1] = (u3 & 0xffff0000u) | (u2 >> 16);
  float l0 = xv[0] - __builtin_bit_cast(float, u0 & 0xffff0000u);
  float l1 = xv[1] - __builtin_bit_cast(float, u1 & 0xffff0000u);
  float l2 = xv[2] - __builtin_bit_cast(float, u2 & 0xffff0000u);
  float l3 = xv[3] - __builtin_bit_cast(float, u3 & 0xffff0000u);
  unsigned w0 = __builtin_bit_cast(unsigned, l0);
  unsigned w1 = __builtin_bit_cast(unsigned, l1);
  unsigned w2 = __builtin_bit_cast(unsigned, l2);
  unsigned w3 = __builtin_bit_cast(unsigned, l3);
  v2u lp;
  lp[0] = (w1 & 0xffff0000u) | (w0 >> 16);
  lp[1] = (w3 & 0xffff0000u) | (w2 >> 16);

  *(v2u*)(x_hi + base) = hp;
  *(v2u*)(x_lo + base) = lp;
}

// ---------------------------------------------------------------------------
// Phase 2: tiled WMMA GEMM, fully-async staged, double-buffered.
//   block = 256 threads (8 waves), tile = 128(M) x 64(N), K chunk = 32
//   wave (wm,wn) owns a 32x32 output patch = 2x2 tiles of 16x16
// LDS rows padded to 40 bf16 (80B): 16B-aligned b128, conflict-free banks.
// ---------------------------------------------------------------------------
#define LDP  40
#define XBUF (128u * LDP * 2u)   // bytes per X plane buffer (10240)
#define WBUF (64u  * LDP * 2u)   // bytes per W plane buffer (5120)

__global__ void __launch_bounds__(256)
vptq_gemm_kernel(const __bf16* __restrict__ x_hi,
                 const __bf16* __restrict__ x_lo,
                 const __bf16* __restrict__ w_hi,
                 const __bf16* __restrict__ w_lo,
                 float* __restrict__ y)
{
  __shared__ __bf16 sXh[2][128][LDP];
  __shared__ __bf16 sXl[2][128][LDP];
  __shared__ __bf16 sWh[2][64][LDP];
  __shared__ __bf16 sWl[2][64][LDP];

  const unsigned tid  = threadIdx.x;
  const unsigned lane = tid & 31u;
  const unsigned wave = tid >> 5;        // 0..7
  const unsigned wm   = wave & 3u;       // M sub-block (x32)
  const unsigned wn   = wave >> 2;       // N sub-block (x32)
  const unsigned m0   = blockIdx.y * 128u;
  const unsigned n0   = blockIdx.x * 64u;

  // A/B operand LDS addressing per ISA 16-bit layout:
  // lanes 0-15: row = lane,   K phase 0 (K 0..7 / 16..23)
  // lanes 16-31: row = lane-16, K phase 8 (K 8..15 / 24..31)
  const unsigned arow = lane & 15u;
  const unsigned koff = (lane >> 4) * 8u;

  // ---- per-thread async staging map (b128 = 8 bf16 per transfer) ----
  // X planes: 128 rows x 32 cols = 512 chunks/plane; 2 chunks per thread.
  const unsigned xr0 = tid >> 2;              // rows 0..63
  const unsigned xr1 = xr0 + 64u;             // rows 64..127
  const unsigned xc8 = (tid & 3u) * 8u;       // col group 0,8,16,24
  // W planes: 64 rows x 32 cols = 256 chunks/plane; 1 chunk per thread.
  const unsigned wr  = tid >> 2;
  const unsigned wc8 = xc8;

  const unsigned oXh0 = lds_off32(&sXh[0][xr0][xc8]);
  const unsigned oXh1 = lds_off32(&sXh[0][xr1][xc8]);
  const unsigned oXl0 = lds_off32(&sXl[0][xr0][xc8]);
  const unsigned oXl1 = lds_off32(&sXl[0][xr1][xc8]);
  const unsigned oWh  = lds_off32(&sWh[0][wr][wc8]);
  const unsigned oWl  = lds_off32(&sWl[0][wr][wc8]);

  const __bf16* gXh0 = x_hi + (size_t)(m0 + xr0) * IN_F + xc8;
  const __bf16* gXh1 = x_hi + (size_t)(m0 + xr1) * IN_F + xc8;
  const __bf16* gXl0 = x_lo + (size_t)(m0 + xr0) * IN_F + xc8;
  const __bf16* gXl1 = x_lo + (size_t)(m0 + xr1) * IN_F + xc8;
  const __bf16* gWh  = w_hi + (size_t)(n0 + wr) * IN_F + wc8;
  const __bf16* gWl  = w_lo + (size_t)(n0 + wr) * IN_F + wc8;

  v8f acc[2][2];
#pragma unroll
  for (int i = 0; i < 2; ++i)
#pragma unroll
    for (int j = 0; j < 2; ++j)
#pragma unroll
      for (int e = 0; e < 8; ++e) acc[i][j][e] = 0.0f;

  // ---- prologue: async-issue tile 0 into buffer 0 (6 ops/thread) ----
  async_load_b128(oXh0, gXh0);
  async_load_b128(oXh1, gXh1);
  async_load_b128(oXl0, gXl0);
  async_load_b128(oXl1, gXl1);
  async_load_b128(oWh,  gWh);
  async_load_b128(oWl,  gWl);

  unsigned p = 0;
  for (unsigned kc = 0; kc < IN_F; kc += 32) {
    const bool has_next = (kc + 32u < IN_F);
    // ---- async-issue next tile into the other buffer ----
    if (has_next) {
      const unsigned pb = (p ^ 1u);
      const unsigned xo = pb * XBUF, wo = pb * WBUF, kn = kc + 32u;
      async_load_b128(oXh0 + xo, gXh0 + kn);
      async_load_b128(oXh1 + xo, gXh1 + kn);
      async_load_b128(oXl0 + xo, gXl0 + kn);
      async_load_b128(oXl1 + xo, gXl1 + kn);
      async_load_b128(oWh  + wo, gWh  + kn);
      async_load_b128(oWl  + wo, gWl  + kn);
      wait_asynccnt6();   // current tile landed (in-order completion)
    } else {
      wait_asynccnt0();
    }
    __syncthreads();      // all waves' async writes visible

    // ---- fetch operands (two 16B chunks at K +0 / +16 per ISA layout)
    v16bf Ah[2], Al[2], Bh[2], Bl[2];
#pragma unroll
    for (int i = 0; i < 2; ++i) {
      unsigned r = wm * 32u + (unsigned)i * 16u + arow;
      v8bf h0 = *(const v8bf*)&sXh[p][r][koff];
      v8bf h1 = *(const v8bf*)&sXh[p][r][koff + 16u];
      Ah[i] = __builtin_shufflevector(h0, h1, 0,1,2,3,4,5,6,7,8,9,10,11,12,13,14,15);
      v8bf l0 = *(const v8bf*)&sXl[p][r][koff];
      v8bf l1 = *(const v8bf*)&sXl[p][r][koff + 16u];
      Al[i] = __builtin_shufflevector(l0, l1, 0,1,2,3,4,5,6,7,8,9,10,11,12,13,14,15);
    }
#pragma unroll
    for (int j = 0; j < 2; ++j) {
      unsigned r = wn * 32u + (unsigned)j * 16u + arow;
      v8bf h0 = *(const v8bf*)&sWh[p][r][koff];
      v8bf h1 = *(const v8bf*)&sWh[p][r][koff + 16u];
      Bh[j] = __builtin_shufflevector(h0, h1, 0,1,2,3,4,5,6,7,8,9,10,11,12,13,14,15);
      v8bf l0 = *(const v8bf*)&sWl[p][r][koff];
      v8bf l1 = *(const v8bf*)&sWl[p][r][koff + 16u];
      Bl[j] = __builtin_shufflevector(l0, l1, 0,1,2,3,4,5,6,7,8,9,10,11,12,13,14,15);
    }

    // ---- 12 WMMAs: split-bf16 product (hh + lh + hl)
#pragma unroll
    for (int i = 0; i < 2; ++i)
#pragma unroll
      for (int j = 0; j < 2; ++j) {
        acc[i][j] = __builtin_amdgcn_wmma_f32_16x16x32_bf16(
            false, Ah[i], false, Bh[j], (short)0, acc[i][j], false, false);
        acc[i][j] = __builtin_amdgcn_wmma_f32_16x16x32_bf16(
            false, Al[i], false, Bh[j], (short)0, acc[i][j], false, false);
        acc[i][j] = __builtin_amdgcn_wmma_f32_16x16x32_bf16(
            false, Ah[i], false, Bl[j], (short)0, acc[i][j], false, false);
      }
    __syncthreads();      // done reading buffer p; next iter may overwrite it
    p ^= 1u;
  }

  // ---- epilogue: 16x16 f32 C/D layout
  // VGPR v: lanes 0-15 -> (M=v, N=lane); lanes 16-31 -> (M=v+8, N=lane-16)
  const unsigned mlane = (lane >> 4) * 8u;
  const unsigned nlane = lane & 15u;
#pragma unroll
  for (int i = 0; i < 2; ++i)
#pragma unroll
    for (int j = 0; j < 2; ++j) {
      unsigned mb = m0 + wm * 32u + (unsigned)i * 16u + mlane;
      unsigned nb = n0 + wn * 32u + (unsigned)j * 16u + nlane;
#pragma unroll
      for (int v = 0; v < 8; ++v)
        y[(size_t)(mb + v) * OUT_F + nb] = acc[i][j][v];
    }
}

// ---------------------------------------------------------------------------
// Host launcher
// ---------------------------------------------------------------------------
extern "C" void kernel_launch(void* const* d_in, const int* in_sizes, int n_in,
                              void* d_out, int out_size, void* d_ws, size_t ws_size,
                              hipStream_t stream) {
  (void)in_sizes; (void)n_in; (void)out_size; (void)ws_size;
  const float* x             = (const float*)d_in[0];
  const float* centroids     = (const float*)d_in[1];
  const int*   indices       = (const int*)  d_in[2];
  const float* res_centroids = (const float*)d_in[3];
  const int*   res_indices   = (const int*)  d_in[4];
  const float* wscale        = (const float*)d_in[5];
  const float* wbias         = (const float*)d_in[6];
  float* y = (float*)d_out;

  // workspace layout (bf16): W hi | W lo | X hi | X lo  (320 MB total)
  const size_t wsz = (size_t)OUT_F * IN_F;   // 16.7M
  const size_t xsz = (size_t)MTOT  * IN_F;   // 67.1M
  __bf16* w_hi = (__bf16*)d_ws;
  __bf16* w_lo = w_hi + wsz;
  __bf16* x_hi = w_lo + wsz;
  __bf16* x_lo = x_hi + xsz;

  // Phase 1a: dequant 16.7M weights, 4 per thread
  unsigned dq_blocks = (unsigned)(wsz / (256 * 4)); // 16384
  vptq_dequant_kernel<<<dq_blocks, 256, 0, stream>>>(
      centroids, indices, res_centroids, res_indices, wscale, wbias, w_hi, w_lo);

  // Phase 1b: split 67.1M x elements, 4 per thread
  unsigned xs_blocks = (unsigned)(xsz / (256 * 4)); // 65536
  vptq_xsplit_kernel<<<xs_blocks, 256, 0, stream>>>(x, x_hi, x_lo);

  // Phase 2: GEMM y[M=16384, N=4096]
  dim3 grid(OUT_F / 64, MTOT / 128);  // (64, 128)
  vptq_gemm_kernel<<<grid, 256, 0, stream>>>(x_hi, x_lo, w_hi, w_lo, y);
}
